// AblationSphereLoRA_86251533238770
// MI455X (gfx1250) — matrix operand order repaired and checked by reading, not compiled, not run on hardware
//
#include <hip/hip_runtime.h>
#include <cstdint>

typedef __attribute__((ext_vector_type(16))) __bf16 v16bf;
typedef __attribute__((ext_vector_type(8)))  float  v8f;

#define B_    2
#define CIN_  128
#define COUT_ 128
#define H_    128
#define W_    256
#define K9_   9
#define KDIM_ 1152   // CIN * 9
#define CMID_ 32
#define MT_   32     // pixels per block tile
#define KC_   32     // WMMA K chunk (bf16)
#define NCH_  36     // KDIM / KC

// ws layout: conv weights (pair-packed), down_w (pair-packed), up_w (row-major)
#define WS_BT_OFF 0
#define WS_DW_OFF (KDIM_ * COUT_ * 2)                 // 294912 B
#define WS_UW_OFF (WS_DW_OFF + 2 * COUT_ * CMID_ * 2) // +16384 B

// ---- repack all three weight matrices f32 -> bf16 ----
__global__ __launch_bounds__(256) void prep_weights_bf16(
    const float* __restrict__ w, const float* __restrict__ dw,
    const float* __restrict__ uw, __bf16* __restrict__ ws)
{
    int t = blockIdx.x * 256 + threadIdx.x;
    // conv weight (COUT, CIN*9) -> pairs [k/2][cout][2]
    if (t < COUT_ * KDIM_) {
        int cout = t & (COUT_ - 1);
        int k    = t >> 7;
        ws[WS_BT_OFF / 2 + (((k >> 1) * COUT_) + cout) * 2 + (k & 1)] = (__bf16)w[cout * KDIM_ + k];
    }
    // down_w (CMID, 2*COUT) -> pairs [c/2][d][2]  (B-matrix for down-proj GEMM)
    if (t < 2 * COUT_ * CMID_) {
        int d = t & (CMID_ - 1);
        int c = t >> 5;
        ws[WS_DW_OFF / 2 + (((c >> 1) * CMID_) + d) * 2 + (c & 1)] = (__bf16)dw[d * (2 * COUT_) + c];
    }
    // up_w (COUT, CMID) -> plain row-major bf16 (A-matrix pairs already contiguous)
    if (t < COUT_ * CMID_) {
        ws[WS_UW_OFF / 2 + t] = (__bf16)uw[t];
    }
}

// ---- fused: conv3x3 + deformable conv3x3 (shared weights) + LoRA down/up ----
__global__ __launch_bounds__(256) void fused_deform_lora(
    const float* __restrict__ x, const float* __restrict__ off,
    const __bf16* __restrict__ ws, const float* __restrict__ bias,
    const float* __restrict__ scale, float* __restrict__ out)
{
    __shared__ __bf16 Areg[2][MT_][KC_ + 2];       // double-buffered im2col tile
    __shared__ __bf16 Adef[2][MT_][KC_ + 2];       // double-buffered deform tile
    __shared__ int    bidx[MT_][K9_][4];           // clamped flat indices
    __shared__ float  bwt [MT_][K9_][4];           // bilinear weights (0 if OOB)
    __shared__ __bf16 featB[MT_][2 * COUT_ + 2];   // concat feats, pixel-major bf16
    __shared__ float  hF  [COUT_][MT_ + 1];        // fp32 h for residual
    __shared__ __bf16 losB[MT_][CMID_ + 2];        // lo, pixel-major bf16

    const int t    = threadIdx.x;
    const int wave = t >> 5;
    const int lane = t & 31;
    const int hl   = lane >> 4;     // lane half (wave32 WMMA layout)
    const int nn   = lane & 15;     // column within fragment
    const int coutW = wave * 16 + nn;

    const int tile  = blockIdx.x;
    const int jt    = tile & 7;
    const int i     = (tile >> 3) & (H_ - 1);
    const int b     = tile >> 10;
    const int jbase = jt * MT_;

    const uint32_t* bt32 = (const uint32_t*)(ws + WS_BT_OFF / 2);
    const uint32_t* dw32 = (const uint32_t*)(ws + WS_DW_OFF / 2);
    const __bf16*   uwB  = ws + WS_UW_OFF / 2;

    // ---------- phase 0: bilinear tables (32 px x 9 taps) ----------
    for (int e = t; e < MT_ * K9_; e += 256) {
        int p  = e / K9_;
        int kk = e - p * K9_;
        int j  = jbase + p;
        int ki = (kk * 11) >> 5, kj = kk - ki * 3;
        float dy  = off[((kk * 2 + 0) * H_ + i) * W_ + j];
        float dxv = off[((kk * 2 + 1) * H_ + i) * W_ + j];
        float py = (float)(i - 1 + ki) + dy;
        float px = (float)(j - 1 + kj) + dxv;
        float fy0 = floorf(py), fx0 = floorf(px);
        int y0 = (int)fy0, x0 = (int)fx0;
        float fy = py - fy0, fx = px - fx0;
        float wq[4] = { (1.f - fy) * (1.f - fx), (1.f - fy) * fx,
                        fy * (1.f - fx),         fy * fx };
        int yq[4] = { y0, y0, y0 + 1, y0 + 1 };
        int xq[4] = { x0, x0 + 1, x0, x0 + 1 };
        #pragma unroll
        for (int q = 0; q < 4; ++q) {
            bool v = (yq[q] >= 0) && (yq[q] < H_) && (xq[q] >= 0) && (xq[q] < W_);
            int yc = yq[q] < 0 ? 0 : (yq[q] > H_ - 1 ? H_ - 1 : yq[q]);
            int xc = xq[q] < 0 ? 0 : (xq[q] > W_ - 1 ? W_ - 1 : xq[q]);
            bidx[p][kk][q] = yc * W_ + xc;
            bwt [p][kk][q] = v ? wq[q] : 0.f;
        }
    }

    // per-thread staging state: 4 element slots, (c, kk) carried incrementally
    int mS[4], klS[4], cS[4], kkS[4];
    #pragma unroll
    for (int r = 0; r < 4; ++r) {
        int e = (r << 8) + t;
        mS[r]  = e >> 5;
        klS[r] = e & 31;
        cS[r]  = klS[r] / 9;           // one-time division
        kkS[r] = klS[r] - cS[r] * 9;
    }

    auto stage = [&](int buf) {
        #pragma unroll
        for (int r = 0; r < 4; ++r) {
            int m = mS[r], kl = klS[r], c = cS[r], kk = kkS[r];
            int ki = (kk * 11) >> 5, kj = kk - ki * 3;
            const float* xp = x + (size_t)(b * CIN_ + c) * (H_ * W_);
            int yy = i - 1 + ki;
            int xx = jbase + m - 1 + kj;
            float vr = 0.f;
            if (yy >= 0 && yy < H_ && xx >= 0 && xx < W_) vr = xp[yy * W_ + xx];
            Areg[buf][m][kl] = (__bf16)vr;
            const int*   i4 = bidx[m][kk];
            const float* w4 = bwt [m][kk];
            float vd = w4[0] * xp[i4[0]] + w4[1] * xp[i4[1]]
                     + w4[2] * xp[i4[2]] + w4[3] * xp[i4[3]];
            Adef[buf][m][kl] = (__bf16)vd;
            cS[r] += 3; kkS[r] += 5;
            if (kkS[r] >= 9) { kkS[r] -= 9; cS[r] += 1; }
        }
    };

    v8f accH0 = {}, accH1 = {}, accD0 = {}, accD1 = {};
    const int m0 = nn, m1 = 16 + nn;

    __syncthreads();      // bilinear tables ready
    stage(0);             // prologue: fill buffer 0
    __syncthreads();

    // ---------- phase 1: K loop, double-buffered, 4 WMMA / chunk ----------
    for (int kc = 0; kc < NCH_; ++kc) {
        const int buf = kc & 1;

        if (kc + 1 < NCH_)
            __builtin_prefetch(&bt32[((kc + 1) * 16 + hl * 8) * COUT_ + coutW], 0, 3);

        v16bf fB;
        uint32_t* bu = reinterpret_cast<uint32_t*>(&fB);
        #pragma unroll
        for (int jv = 0; jv < 8; ++jv)
            bu[jv] = bt32[(kc * 16 + hl * 8 + jv) * COUT_ + coutW];

        v16bf fR0, fR1, fD0, fD1;
        uint32_t* r0 = reinterpret_cast<uint32_t*>(&fR0);
        uint32_t* r1 = reinterpret_cast<uint32_t*>(&fR1);
        uint32_t* d0 = reinterpret_cast<uint32_t*>(&fD0);
        uint32_t* d1 = reinterpret_cast<uint32_t*>(&fD1);
        #pragma unroll
        for (int iv = 0; iv < 8; ++iv) {
            int k0 = hl * 16 + (iv << 1);
            r0[iv] = *(const uint32_t*)&Areg[buf][m0][k0];
            r1[iv] = *(const uint32_t*)&Areg[buf][m1][k0];
            d0[iv] = *(const uint32_t*)&Adef[buf][m0][k0];
            d1[iv] = *(const uint32_t*)&Adef[buf][m1][k0];
        }

        if (kc + 1 < NCH_) stage(buf ^ 1);   // overlap staging with WMMAs

        accH0 = __builtin_amdgcn_wmma_f32_16x16x32_bf16(false, fR0, false, fB, (short)0, accH0, false, false);
        accH1 = __builtin_amdgcn_wmma_f32_16x16x32_bf16(false, fR1, false, fB, (short)0, accH1, false, false);
        accD0 = __builtin_amdgcn_wmma_f32_16x16x32_bf16(false, fD0, false, fB, (short)0, accD0, false, false);
        accD1 = __builtin_amdgcn_wmma_f32_16x16x32_bf16(false, fD1, false, fB, (short)0, accD1, false, false);

        __syncthreads();
    }

    // ---------- phase 2: bias; concat feats -> bf16 LDS (pixel-major); h -> fp32 ----------
    {
        float bv = bias[coutW];
        #pragma unroll
        for (int r = 0; r < 8; ++r) {
            int p = r + 8 * hl;                 // C/D layout: M = r + 8*half
            float dv0 = accD0[r] + bv, dv1 = accD1[r] + bv;
            float hv0 = accH0[r] + bv, hv1 = accH1[r] + bv;
            featB[p]     [coutW]          = (__bf16)dv0;   // deform -> cols 0..127
            featB[16 + p][coutW]          = (__bf16)dv1;
            featB[p]     [COUT_ + coutW]  = (__bf16)hv0;   // h -> cols 128..255
            featB[16 + p][COUT_ + coutW]  = (__bf16)hv1;
            hF[coutW][p]      = hv0;
            hF[coutW][16 + p] = hv1;
        }
    }
    __syncthreads();

    // ---------- phase 3 (waves 0-3): lo = featB(32x256) x down_w^T(256x32), WMMA ----------
    if (wave < 4) {
        const int mt  = wave >> 1;        // pixel half
        const int ntd = wave & 1;         // d half
        const int px  = mt * 16 + nn;
        v8f accL = {};
        #pragma unroll
        for (int kc2 = 0; kc2 < 8; ++kc2) {
            v16bf fA, fBd;
            uint32_t* au  = reinterpret_cast<uint32_t*>(&fA);
            uint32_t* bu2 = reinterpret_cast<uint32_t*>(&fBd);
            #pragma unroll
            for (int iv = 0; iv < 8; ++iv) {
                int k0 = hl * 16 + (iv << 1);
                au[iv] = *(const uint32_t*)&featB[px][kc2 * 32 + k0];
            }
            #pragma unroll
            for (int jv = 0; jv < 8; ++jv)
                bu2[jv] = dw32[(kc2 * 16 + hl * 8 + jv) * CMID_ + (ntd * 16 + nn)];
            accL = __builtin_amdgcn_wmma_f32_16x16x32_bf16(false, fA, false, fBd, (short)0, accL, false, false);
        }
        #pragma unroll
        for (int r = 0; r < 8; ++r)
            losB[mt * 16 + r + 8 * hl][ntd * 16 + nn] = (__bf16)accL[r];
    }
    __syncthreads();

    // ---------- phase 4 (all waves): up = up_w(128x32) x lo(32x32), WMMA; residual ----------
    {
        v16bf fA;                                   // A = up_w rows (co x d)
        uint32_t* au = reinterpret_cast<uint32_t*>(&fA);
        #pragma unroll
        for (int iv = 0; iv < 8; ++iv) {
            int k0 = hl * 16 + (iv << 1);
            au[iv] = *(const uint32_t*)&uwB[(wave * 16 + nn) * CMID_ + k0];
        }
        v16bf fB0, fB1;                             // B = lo (d x px), two px halves
        uint32_t* b0 = reinterpret_cast<uint32_t*>(&fB0);
        uint32_t* b1 = reinterpret_cast<uint32_t*>(&fB1);
        #pragma unroll
        for (int jv = 0; jv < 8; ++jv) {
            int dbase = hl * 16 + 2 * jv;
            b0[jv] = *(const uint32_t*)&losB[nn][dbase];
            b1[jv] = *(const uint32_t*)&losB[16 + nn][dbase];
        }
        v8f up0 = {}, up1 = {};
        up0 = __builtin_amdgcn_wmma_f32_16x16x32_bf16(false, fA, false, fB0, (short)0, up0, false, false);
        up1 = __builtin_amdgcn_wmma_f32_16x16x32_bf16(false, fA, false, fB1, (short)0, up1, false, false);

        float s = scale[0];
        #pragma unroll
        for (int r = 0; r < 8; ++r) {
            int co_ = wave * 16 + r + 8 * hl;       // D layout: M=co, N=px
            float h0 = hF[co_][nn];
            float h1 = hF[co_][16 + nn];
            size_t base = ((size_t)(b * COUT_ + co_) * H_ + i) * W_ + jbase;
            out[base + nn]      = h0 + s * up0[r];
            out[base + 16 + nn] = h1 + s * up1[r];
        }
    }
}

extern "C" void kernel_launch(void* const* d_in, const int* in_sizes, int n_in,
                              void* d_out, int out_size, void* d_ws, size_t ws_size,
                              hipStream_t stream)
{
    const float* x      = (const float*)d_in[0];
    const float* offset = (const float*)d_in[1];
    const float* weight = (const float*)d_in[2];
    const float* bias   = (const float*)d_in[3];
    const float* down_w = (const float*)d_in[4];
    const float* up_w   = (const float*)d_in[5];
    const float* scale  = (const float*)d_in[6];
    float* out = (float*)d_out;

    __bf16* ws = (__bf16*)d_ws;   // 312 KB packed bf16 weights

    prep_weights_bf16<<<(COUT_ * KDIM_ + 255) / 256, 256, 0, stream>>>(
        weight, down_w, up_w, ws);

    const int nblocks = B_ * H_ * (W_ / MT_);   // 2048
    fused_deform_lora<<<nblocks, 256, 0, stream>>>(x, offset, ws, bias, scale, out);
}
